// Pips_18726057410750
// MI455X (gfx1250) — matrix operand order repaired and checked by reading, not compiled, not run on hardware
//
#include <hip/hip_runtime.h>
#include <hip/hip_bf16.h>
#include <math.h>

#define B_      2
#define S_      8
#define N_      256
#define H_      384
#define W_      512
#define STRIDE_ 8
#define LATENT_ 128
#define RADIUS_ 3
#define HS_     48
#define WS_     64
#define HW_     (HS_*WS_)   /* 3072 */
#define BS_     (B_*S_)     /* 16   */
#define KWIN_   49
#define LRR_    196
#define DIN_    327
#define DOUT_   130
#define ITERS_  3

typedef __attribute__((ext_vector_type(16))) __bf16 bf16x16;
typedef __attribute__((ext_vector_type(8)))  __bf16 bf16x8;
typedef __attribute__((ext_vector_type(8)))  float  f32x8;

__device__ __forceinline__ unsigned short f2bf(float x) {
    unsigned int u = __float_as_uint(x);
    u += 0x7FFFu + ((u >> 16) & 1u);          // round-to-nearest-even
    return (unsigned short)(u >> 16);
}

// ---------------------------------------------------------------------------
// 1) Encoder: stride-8 8x8 conv == per-patch GEMM (fp32 scalar; 2.4 GFLOP)
// ---------------------------------------------------------------------------
__global__ void k_encoder(const float* __restrict__ rgbs, const float* __restrict__ We,
                          const float* __restrict__ be, float* __restrict__ fmaps) {
    int idx = blockIdx.x * blockDim.x + threadIdx.x;
    if (idx >= BS_ * HW_ * LATENT_) return;
    int c  = idx % LATENT_;
    int hw = (idx / LATENT_) % HW_;
    int bs = idx / (LATENT_ * HW_);
    int h0 = (hw / WS_) * STRIDE_, w0 = (hw % WS_) * STRIDE_;
    const float* img = rgbs + (long)bs * 3 * H_ * W_;
    const float* wt  = We + c * 192;
    float acc = be[c];
    #pragma unroll
    for (int ci = 0; ci < 3; ci++)
        #pragma unroll
        for (int ky = 0; ky < 8; ky++) {
            const float* row = img + ci * H_ * W_ + (h0 + ky) * W_ + w0;
            const float* wr  = wt + ci * 64 + ky * 8;
            #pragma unroll
            for (int kx = 0; kx < 8; kx++) acc += row[kx] * wr[kx];
        }
    fmaps[((long)bs * LATENT_ + c) * HW_ + hw] = acc;
}

// Transpose fmaps (bs,C,HW) -> bf16 (bs,HW,C) so WMMA-B fragments are contiguous.
__global__ void k_fmT(const float* __restrict__ fmaps, unsigned short* __restrict__ fmT) {
    int idx = blockIdx.x * blockDim.x + threadIdx.x;
    if (idx >= BS_ * HW_ * LATENT_) return;
    int c  = idx % LATENT_;
    int hw = (idx / LATENT_) % HW_;
    int bs = idx / (LATENT_ * HW_);
    fmT[((long)bs * HW_ + hw) * LATENT_ + c] = f2bf(fmaps[((long)bs * LATENT_ + c) * HW_ + hw]);
}

// ---------------------------------------------------------------------------
// 2) Init: coords = xys/8 broadcast over s; preds2 slots 0,1; cf0 bilinear sample
// ---------------------------------------------------------------------------
__global__ void k_init_coords(const float* __restrict__ xys, float* __restrict__ coords,
                              float* __restrict__ preds2) {
    int idx = blockIdx.x * blockDim.x + threadIdx.x;
    if (idx >= B_ * S_ * N_) return;
    int n = idx % N_, s = (idx / N_) % S_, b = idx / (N_ * S_);
    float x = xys[(b * N_ + n) * 2 + 0];
    float y = xys[(b * N_ + n) * 2 + 1];
    int pn = (b * S_ + s) * N_ + n;
    coords[pn * 2 + 0] = x / 8.0f;
    coords[pn * 2 + 1] = y / 8.0f;
    #pragma unroll
    for (int slot = 0; slot < 2; slot++) {
        long o = (((long)slot * B_ + b) * S_ + s) * N_ * 2 + n * 2;
        preds2[o] = x; preds2[o + 1] = y;
    }
}

__global__ void k_init_cf(const float* __restrict__ xys, const float* __restrict__ fmaps,
                          float* __restrict__ cf, unsigned short* __restrict__ cfb) {
    int idx = blockIdx.x * blockDim.x + threadIdx.x;
    if (idx >= B_ * N_ * LATENT_) return;
    int c = idx % LATENT_, n = (idx / LATENT_) % N_, b = idx / (LATENT_ * N_);
    float X = xys[(b * N_ + n) * 2 + 0] / 8.0f;
    float Y = xys[(b * N_ + n) * 2 + 1] / 8.0f;
    float X0 = floorf(X), Y0 = floorf(Y);
    int x0 = min(max((int)X0, 0), WS_ - 1), x1 = min(max((int)X0 + 1, 0), WS_ - 1);
    int y0 = min(max((int)Y0, 0), HS_ - 1), y1 = min(max((int)Y0 + 1, 0), HS_ - 1);
    const float* fm = fmaps + ((long)(b * S_ + 0) * LATENT_ + c) * HW_;
    float f00 = fm[y0 * WS_ + x0], f01 = fm[y0 * WS_ + x1];
    float f10 = fm[y1 * WS_ + x0], f11 = fm[y1 * WS_ + x1];
    float wx1 = X - X0, wx0 = 1.0f - wx1, wy1 = Y - Y0, wy0 = 1.0f - wy1;
    float v = wx0 * wy0 * f00 + wx1 * wy0 * f01 + wx0 * wy1 * f10 + wx1 * wy1 * f11;
    unsigned short vb = f2bf(v);
    for (int s = 0; s < S_; s++) {
        long o = ((long)(b * S_ + s) * N_ + n) * LATENT_ + c;
        cf[o] = v; cfb[o] = vb;
    }
}

// ---------------------------------------------------------------------------
// 3) corr = cf (N x C) x fmapsT (C x HW) / sqrt(C), per (b,s): bf16 WMMA, f32 acc.
//    The 64-col x K=128 B panel is DMAed to LDS once per block with
//    GLOBAL_LOAD_ASYNC_TO_LDS_B128 (ASYNCcnt), shared by all 8 waves (8x less
//    L2 traffic), and fed to WMMA via ds_load — the per-WMMA wait becomes a
//    short DS wait instead of a global-load wait.
//    A lanes: lane<16 -> K {kb..kb+7, kb+16..kb+23}; lane>=16 -> +8 / +24.
//    B lanes: same K mapping as A.
// ---------------------------------------------------------------------------
#define BPAD_ 136   /* row stride in ushorts: 272 B => lanes hit distinct bank groups */

__global__ __launch_bounds__(256) void k_corr_wmma(const unsigned short* __restrict__ cfb_,
                                                   const unsigned short* __restrict__ fmT_,
                                                   float* __restrict__ corr0) {
    __shared__ unsigned short smemB[64][BPAD_];      // 64 cols x 128 K (+pad) = 17 KB
    const __bf16* cfb = (const __bf16*)cfb_;
    int tid  = threadIdx.x;
    int wave = tid >> 5, lane = tid & 31;
    int mt = blockIdx.y * 8 + wave;        // 16 M-tiles cover N_=256 points
    int colBase = blockIdx.x * 64;         // 48 column groups cover HW_=3072
    int bs = blockIdx.z;
    bool lo = lane < 16;
    int row = mt * 16 + (lane & 15);

    // --- async DMA: stage B panel (64 cols x 128 K bf16) into LDS -----------
    unsigned ldsBase = (unsigned)(unsigned long long)(uintptr_t)&smemB[0][0];
    #pragma unroll
    for (int q = 0; q < 4; q++) {
        int chunk = tid + 256 * q;         // 1024 x 16-byte chunks
        int col = chunk >> 4;              // 16 chunks per column
        int k16 = chunk & 15;              // chunk = 8 bf16 K-values
        const unsigned short* gp =
            fmT_ + ((size_t)bs * HW_ + colBase + col) * LATENT_ + k16 * 8;
        unsigned ldsoff = ldsBase + (unsigned)(col * BPAD_ + k16 * 8) * 2u;
        unsigned long long gaddr = (unsigned long long)(uintptr_t)gp;
        asm volatile("global_load_async_to_lds_b128 %0, %1, off"
                     :: "v"(ldsoff), "v"(gaddr) : "memory");
    }
    asm volatile("s_wait_asynccnt 0x0" ::: "memory");
    __syncthreads();

    f32x8 acc[4];
    f32x8 z = {0.f, 0.f, 0.f, 0.f, 0.f, 0.f, 0.f, 0.f};
    #pragma unroll
    for (int j = 0; j < 4; j++) acc[j] = z;

    const __bf16* abase = cfb + ((long)bs * N_ + row) * LATENT_;

    #pragma unroll
    for (int kb = 0; kb < 128; kb += 32) {
        const __bf16* ap = abase + kb + (lo ? 0 : 8);
        bf16x8 a0 = *(const bf16x8*)ap;
        bf16x8 a1 = *(const bf16x8*)(ap + 16);
        bf16x16 A;
        #pragma unroll
        for (int i = 0; i < 8; i++) { A[i] = a0[i]; A[8 + i] = a1[i]; }
        if (kb < 96) __builtin_prefetch(abase + kb + 32, 0, 0);
        #pragma unroll
        for (int j = 0; j < 4; j++) {
            const unsigned short* bp = &smemB[j * 16 + (lane & 15)][kb + (lo ? 0 : 8)];
            bf16x8 b0 = *(const bf16x8*)(const void*)bp;        // ds_load_b128
            bf16x8 b1 = *(const bf16x8*)(const void*)(bp + 16); // ds_load_b128
            bf16x16 Bm;
            #pragma unroll
            for (int i = 0; i < 8; i++) { Bm[i] = b0[i]; Bm[8 + i] = b1[i]; }
            acc[j] = __builtin_amdgcn_wmma_f32_16x16x32_bf16(
                false, A, false, Bm, (short)0, acc[j], false, false);
        }
    }
    const float scale = 0.08838834764831845f;  // 1/sqrt(128)
    #pragma unroll
    for (int j = 0; j < 4; j++) {
        int ncol = colBase + j * 16 + (lane & 15);
        float* op = corr0 + (long)bs * N_ * HW_ + ncol;
        #pragma unroll
        for (int r = 0; r < 8; r++) {
            int m = mt * 16 + r + (lo ? 0 : 8);
            op[(long)m * HW_] = acc[j][r] * scale;
        }
    }
}

// ---------------------------------------------------------------------------
// 4) pool2x2 over (BS*N, Hi, Wi)
// ---------------------------------------------------------------------------
__global__ void k_pool(const float* __restrict__ in, float* __restrict__ out, int Hi, int Wi) {
    int Ho = Hi / 2, Wo = Wi / 2;
    int idx = blockIdx.x * blockDim.x + threadIdx.x;
    int total = BS_ * N_ * Ho * Wo;
    if (idx >= total) return;
    int wo = idx % Wo, ho = (idx / Wo) % Ho;
    long img = idx / (Ho * Wo);
    const float* p = in + img * Hi * Wi + (2 * ho) * Wi + 2 * wo;
    out[idx] = 0.25f * (p[0] + p[1] + p[Wi] + p[Wi + 1]);
}

__device__ __forceinline__ float bil_ac(const float* img, int Hl, int Wl, float ys, float xs) {
    int y0 = (int)floorf(ys); int y1 = min(y0 + 1, Hl - 1);
    int x0 = (int)floorf(xs); int x1 = min(x0 + 1, Wl - 1);
    float wy = ys - (float)y0, wx = xs - (float)x0;
    float r0 = img[y0 * Wl + x0] * (1.f - wx) + img[y0 * Wl + x1] * wx;
    float r1 = img[y1 * Wl + x0] * (1.f - wx) + img[y1 * Wl + x1] * wx;
    return r0 * (1.f - wy) + r1 * wy;
}

// 5) fused 4-level align-corners resize-accumulate -> features_correlations[it]
__global__ void k_fc(const float* __restrict__ c0, const float* __restrict__ c1,
                     const float* __restrict__ c2, const float* __restrict__ c3,
                     float* __restrict__ fc, int it) {
    int idx = blockIdx.x * blockDim.x + threadIdx.x;
    if (idx >= BS_ * N_ * HW_) return;
    int w = idx % WS_, h = (idx / WS_) % HS_;
    long img = idx / HW_;
    int bs = (int)(img / N_), n = (int)(img % N_);
    float v = c0[img * HW_ + h * WS_ + w];                                 // level 0: identity
    v += bil_ac(c1 + img * 768, 24, 32, h * (23.0f / 47.0f), w * (31.0f / 63.0f));
    v += bil_ac(c2 + img * 192, 12, 16, h * (11.0f / 47.0f), w * (15.0f / 63.0f));
    v += bil_ac(c3 + img * 48,   6,  8, h * (5.0f  / 47.0f), w * (7.0f  / 63.0f));
    fc[((((long)bs * ITERS_ + it) * N_ + n) * HS_ + h) * WS_ + w] = v;
}

// 6) 49-point bilinear pyramid sampler -> corrs (bs, n, 196)
__global__ void k_sample(const float* __restrict__ c0, const float* __restrict__ c1,
                         const float* __restrict__ c2, const float* __restrict__ c3,
                         const float* __restrict__ coords, float* __restrict__ corrs) {
    int idx = blockIdx.x * blockDim.x + threadIdx.x;
    if (idx >= BS_ * N_ * LRR_) return;
    int kid = idx % KWIN_;
    int lvl = (idx / KWIN_) & 3;
    int pn  = idx / LRR_;
    float x = coords[pn * 2 + 0], y = coords[pn * 2 + 1];
    float inv = 1.0f / (float)(1 << lvl);
    float px = x * inv + (float)(kid % 7 - RADIUS_);
    float py = y * inv + (float)(kid / 7 - RADIUS_);
    int Hl = HS_ >> lvl, Wl = WS_ >> lvl;
    const float* img;
    switch (lvl) { case 0: img = c0; break; case 1: img = c1; break;
                   case 2: img = c2; break; default: img = c3; }
    img += (long)pn * Hl * Wl;
    float X0 = floorf(px), Y0 = floorf(py);
    int x0 = min(max((int)X0, 0), Wl - 1), x1 = min(max((int)X0 + 1, 0), Wl - 1);
    int y0 = min(max((int)Y0, 0), Hl - 1), y1 = min(max((int)Y0 + 1, 0), Hl - 1);
    float wx1 = px - X0, wx0 = 1.0f - wx1, wy1 = py - Y0, wy0 = 1.0f - wy1;
    float val = wx0 * wy0 * img[y0 * Wl + x0] + wx1 * wy0 * img[y0 * Wl + x1]
              + wx0 * wy1 * img[y1 * Wl + x0] + wx1 * wy1 * img[y1 * Wl + x1];
    corrs[(long)pn * LRR_ + lvl * KWIN_ + kid] = val;
}

// 7) delta = [cf | corrs | flows] @ Wd + bd   (4096 x 327 x 130, scalar)
__global__ void k_delta(const float* __restrict__ cf, const float* __restrict__ corrs,
                        const float* __restrict__ coords, const float* __restrict__ Wd,
                        const float* __restrict__ bd, float* __restrict__ delta) {
    int idx = blockIdx.x * blockDim.x + threadIdx.x;
    if (idx >= B_ * N_ * S_ * DOUT_) return;
    int col = idx % DOUT_;
    int row = idx / DOUT_;
    int s = row % S_, n = (row / S_) % N_, b = row / (S_ * N_);
    int pn = (b * S_ + s) * N_ + n;
    float acc = bd[col];
    const float* cfr = cf + (long)pn * LATENT_;
    for (int k = 0; k < LATENT_; k++) acc += cfr[k] * Wd[k * DOUT_ + col];
    const float* cr = corrs + (long)pn * LRR_;
    for (int k = 0; k < LRR_; k++) acc += cr[k] * Wd[(LATENT_ + k) * DOUT_ + col];
    int p0 = (b * S_ + 0) * N_ + n;
    float fx = coords[pn * 2 + 0] - coords[p0 * 2 + 0];
    float fy = coords[pn * 2 + 1] - coords[p0 * 2 + 1];
    float t  = s * (8.0f / 7.0f);                         // linspace(0, 8, 8)
    acc += fx * Wd[(DIN_ - 3) * DOUT_ + col] + fy * Wd[(DIN_ - 2) * DOUT_ + col]
         + t  * Wd[(DIN_ - 1) * DOUT_ + col];
    delta[(long)row * DOUT_ + col] = acc;
}

// 8) coordinate update + preds / preds2 writes
__global__ void k_coord_update(const float* __restrict__ delta, float* __restrict__ coords,
                               float* __restrict__ preds, float* __restrict__ preds2,
                               int it, int last) {
    int idx = blockIdx.x * blockDim.x + threadIdx.x;
    if (idx >= B_ * S_ * N_) return;
    int n = idx % N_, s = (idx / N_) % S_, b = idx / (N_ * S_);
    long row = ((long)(b * N_ + n)) * S_ + s;
    int pn = (b * S_ + s) * N_ + n;
    float x = coords[pn * 2 + 0] + delta[row * DOUT_ + 0];
    float y = coords[pn * 2 + 1] + delta[row * DOUT_ + 1];
    coords[pn * 2 + 0] = x; coords[pn * 2 + 1] = y;
    float px = x * 8.0f, py = y * 8.0f;
    long p = (((long)it * B_ + b) * S_ + s) * N_ * 2 + n * 2;
    preds[p] = px; preds[p + 1] = py;
    long q = (((long)(2 + it) * B_ + b) * S_ + s) * N_ * 2 + n * 2;
    preds2[q] = px; preds2[q + 1] = py;
    if (last) {
        for (int slot = 5; slot < 7; slot++) {
            long r = (((long)slot * B_ + b) * S_ + s) * N_ * 2 + n * 2;
            preds2[r] = px; preds2[r + 1] = py;
        }
    }
}

// 9) dfeat -> LN -> LN -> @Wu + bu -> gelu -> += cf (one block per row)
__global__ __launch_bounds__(128) void k_feat_update(const float* __restrict__ delta,
        const float* __restrict__ ln_g, const float* __restrict__ ln_b,
        const float* __restrict__ Wu, const float* __restrict__ bu,
        float* __restrict__ cf, unsigned short* __restrict__ cfb) {
    __shared__ float red[LATENT_];
    __shared__ float sh[LATENT_];
    int j = threadIdx.x;
    int r = blockIdx.x;
    int s = r % S_, n = (r / S_) % N_, b = r / (S_ * N_);
    float x = delta[(long)r * DOUT_ + 2 + j];
    #pragma unroll
    for (int pass = 0; pass < 2; pass++) {               // reference applies LN twice
        red[j] = x; __syncthreads();
        for (int off = 64; off > 0; off >>= 1) { if (j < off) red[j] += red[j + off]; __syncthreads(); }
        float mu = red[0] * (1.0f / LATENT_); __syncthreads();
        float dm = x - mu;
        red[j] = dm * dm; __syncthreads();
        for (int off = 64; off > 0; off >>= 1) { if (j < off) red[j] += red[j + off]; __syncthreads(); }
        float var = red[0] * (1.0f / LATENT_); __syncthreads();
        x = dm * rsqrtf(var + 1e-5f) * ln_g[j] + ln_b[j];
    }
    sh[j] = x; __syncthreads();
    float acc = bu[j];
    for (int k = 0; k < LATENT_; k++) acc += sh[k] * Wu[k * LATENT_ + j];
    float g = 0.5f * acc * (1.0f + erff(acc * 0.7071067811865476f));  // exact gelu
    long o = ((long)((b * S_ + s) * N_ + n)) * LATENT_ + j;
    float nv = g + cf[o];
    cf[o] = nv; cfb[o] = f2bf(nv);
}

// 10) vis = cf @ Wv + bv
__global__ void k_vis(const float* __restrict__ cf, const float* __restrict__ Wv,
                      const float* __restrict__ bv, float* __restrict__ vis) {
    int idx = blockIdx.x * blockDim.x + threadIdx.x;
    if (idx >= B_ * S_ * N_) return;
    const float* row = cf + (long)idx * LATENT_;
    float acc = bv[0];
    for (int k = 0; k < LATENT_; k++) acc += row[k] * Wv[k];
    vis[idx] = acc;
}

// ---------------------------------------------------------------------------
extern "C" void kernel_launch(void* const* d_in, const int* in_sizes, int n_in,
                              void* d_out, int out_size, void* d_ws, size_t ws_size,
                              hipStream_t stream) {
    const float* xys  = (const float*)d_in[0];
    const float* rgbs = (const float*)d_in[1];
    const float* We   = (const float*)d_in[2];
    const float* be   = (const float*)d_in[3];
    const float* Wd   = (const float*)d_in[4];
    const float* bd   = (const float*)d_in[5];
    const float* ln_g = (const float*)d_in[6];
    const float* ln_b = (const float*)d_in[7];
    const float* Wu   = (const float*)d_in[8];
    const float* bu   = (const float*)d_in[9];
    const float* Wv   = (const float*)d_in[10];
    const float* bv   = (const float*)d_in[11];
    (void)in_sizes; (void)n_in; (void)out_size; (void)ws_size;   // iterations fixed = 3

    char* wsp = (char*)d_ws;
    size_t off = 0;
    auto carve = [&](size_t bytes) -> char* {
        char* p = wsp + off; off += (bytes + 255) & ~(size_t)255; return p;
    };
    float*          fmaps = (float*)         carve((size_t)BS_ * LATENT_ * HW_ * 4);
    unsigned short* fmT   = (unsigned short*)carve((size_t)BS_ * HW_ * LATENT_ * 2);
    float*          cf    = (float*)         carve((size_t)BS_ * N_ * LATENT_ * 4);
    unsigned short* cfb   = (unsigned short*)carve((size_t)BS_ * N_ * LATENT_ * 2);
    float*          coords= (float*)         carve((size_t)BS_ * N_ * 2 * 4);
    float*          corr0 = (float*)         carve((size_t)BS_ * N_ * HW_ * 4);
    float*          corr1 = (float*)         carve((size_t)BS_ * N_ * (HW_/4) * 4);
    float*          corr2 = (float*)         carve((size_t)BS_ * N_ * (HW_/16) * 4);
    float*          corr3 = (float*)         carve((size_t)BS_ * N_ * (HW_/64) * 4);
    float*          corrs = (float*)         carve((size_t)BS_ * N_ * LRR_ * 4);
    float*          delta = (float*)         carve((size_t)B_ * N_ * S_ * DOUT_ * 4);

    float* out    = (float*)d_out;
    float* preds  = out;
    float* preds2 = out + (size_t)ITERS_ * B_ * S_ * N_ * 2;
    float* vis    = preds2 + (size_t)7 * B_ * S_ * N_ * 2;
    float* fc     = vis + (size_t)B_ * S_ * N_;

    {   // encoder + transpose-to-bf16
        int total = BS_ * HW_ * LATENT_;
        k_encoder<<<(total + 255) / 256, 256, 0, stream>>>(rgbs, We, be, fmaps);
        k_fmT    <<<(total + 255) / 256, 256, 0, stream>>>(fmaps, fmT);
    }
    {   // initial coords / preds2[0..1] / cf0
        int t = B_ * S_ * N_;
        k_init_coords<<<(t + 255) / 256, 256, 0, stream>>>(xys, coords, preds2);
        int t2 = B_ * N_ * LATENT_;
        k_init_cf<<<(t2 + 255) / 256, 256, 0, stream>>>(xys, fmaps, cf, cfb);
    }

    for (int it = 0; it < ITERS_; it++) {
        k_corr_wmma<<<dim3(HW_ / 64, 2, BS_), 256, 0, stream>>>(cfb, fmT, corr0);

        int t1 = BS_ * N_ * (HW_ / 4);
        k_pool<<<(t1 + 255) / 256, 256, 0, stream>>>(corr0, corr1, HS_,     WS_);
        int t2 = BS_ * N_ * (HW_ / 16);
        k_pool<<<(t2 + 255) / 256, 256, 0, stream>>>(corr1, corr2, HS_ / 2, WS_ / 2);
        int t3 = BS_ * N_ * (HW_ / 64);
        k_pool<<<(t3 + 255) / 256, 256, 0, stream>>>(corr2, corr3, HS_ / 4, WS_ / 4);

        int tf = BS_ * N_ * HW_;
        k_fc<<<(tf + 255) / 256, 256, 0, stream>>>(corr0, corr1, corr2, corr3, fc, it);

        int ts = BS_ * N_ * LRR_;
        k_sample<<<(ts + 255) / 256, 256, 0, stream>>>(corr0, corr1, corr2, corr3, coords, corrs);

        int td = B_ * N_ * S_ * DOUT_;
        k_delta<<<(td + 255) / 256, 256, 0, stream>>>(cf, corrs, coords, Wd, bd, delta);

        int tc = B_ * S_ * N_;
        k_coord_update<<<(tc + 255) / 256, 256, 0, stream>>>(delta, coords, preds, preds2,
                                                             it, it == ITERS_ - 1);

        k_feat_update<<<B_ * N_ * S_, 128, 0, stream>>>(delta, ln_g, ln_b, Wu, bu, cf, cfb);
    }

    int tv = B_ * S_ * N_;
    k_vis<<<(tv + 255) / 256, 256, 0, stream>>>(cf, Wv, bv, vis);
}
